// MultiHeadLatentAttn_42606075576437
// MI455X (gfx1250) — matrix operand order repaired and checked by reading, not compiled, not run on hardware
//
#include <hip/hip_runtime.h>

#define SEQ   2048
#define BATCH 2
#define MDIM  2048
#define L3    1536   // 3*LATENT_DIM
#define LAT   512
#define UPD   6144   // 3*MODEL_DIM
#define PDIM  1024
#define NHD   16
#define HD    128
#define PHD   64
#define ROWS  (BATCH*SEQ)
#define RTHETA 50000.0f

typedef __attribute__((ext_vector_type(16))) __bf16       v16bf;
typedef __attribute__((ext_vector_type(8)))  float        v8f;
typedef __attribute__((ext_vector_type(4)))  unsigned int u32x4;

union F16x16 { u32x4 u[2]; v16bf v; };

__device__ __forceinline__ unsigned short f2bf(float f) {
    unsigned int u = __float_as_uint(f);
    u += 0x7FFFu + ((u >> 16) & 1u);           // round to nearest even
    return (unsigned short)(u >> 16);
}
__device__ __forceinline__ float bf2f(unsigned short h) {
    return __uint_as_float(((unsigned int)h) << 16);
}
__device__ __forceinline__ v8f wmma_bf16(v16bf a, v16bf b, v8f c) {
    return __builtin_amdgcn_wmma_f32_16x16x32_bf16(false, a, false, b, (short)0, c, false, false);
}

// ---------------- conversion kernels ----------------
__global__ void k_cvt(const float* __restrict__ src, unsigned short* __restrict__ dst, size_t n) {
    size_t i = (size_t)blockIdx.x * blockDim.x + threadIdx.x;
    if (i < n) dst[i] = f2bf(src[i]);
}

// src is [K][N] row-major fp32 ; dst is [N][K] bf16 (B^T layout for WMMA B-fragments)
__global__ void k_cvt_tr(const float* __restrict__ src, unsigned short* __restrict__ dst,
                         int K, int N) {
    size_t i = (size_t)blockIdx.x * blockDim.x + threadIdx.x;
    if (i < (size_t)K * N) {
        int k = (int)(i / N);
        int n = (int)(i % N);
        dst[(size_t)n * K + k] = f2bf(src[i]);
    }
}

// ---------------- bf16 WMMA GEMM, 64x64 per wave (main path) ----------------
// C[M,N] = A[M,K](bf16,row-major) * Bt[N,K](bf16) + bias
// block = 256x128 (8 waves in 4x2), wave = 64x64 = 4x4 WMMA subtiles
// 16 WMMA per 16 fragment loads -> 16 B loaded per WMMA (2x the 32x32 version)
__global__ __launch_bounds__(256)
void k_gemm_bf16_big(const unsigned short* __restrict__ A, int lda,
                     const unsigned short* __restrict__ Bt, int ldb,
                     const float* __restrict__ bias,
                     float* __restrict__ Cf, unsigned short* __restrict__ Cb,
                     int ldc, int K) {
    const int lane = threadIdx.x & 31;
    const int w    = threadIdx.x >> 5;
    const int lm   = lane & 15;
    const int lh   = lane >> 4;
    const int m0   = blockIdx.x * 256 + (w & 3) * 64;
    const int n0   = blockIdx.y * 128 + (w >> 2) * 64;

    const v8f vz = {0.f,0.f,0.f,0.f,0.f,0.f,0.f,0.f};
    v8f acc[4][4];
    for (int i = 0; i < 4; ++i)
        for (int j = 0; j < 4; ++j) acc[i][j] = vz;

    for (int k0 = 0; k0 < K; k0 += 32) {
        F16x16 af[4], bf_[4];
        for (int i = 0; i < 4; ++i) {
            const u32x4* p = (const u32x4*)(A + (size_t)(m0 + i * 16 + lm) * lda + k0 + lh * 8);
            af[i].u[0] = p[0]; af[i].u[1] = p[2];
        }
        for (int j = 0; j < 4; ++j) {
            const u32x4* p = (const u32x4*)(Bt + (size_t)(n0 + j * 16 + lm) * ldb + k0 + lh * 8);
            bf_[j].u[0] = p[0]; bf_[j].u[1] = p[2];
        }
        for (int i = 0; i < 4; ++i)
            for (int j = 0; j < 4; ++j)
                acc[i][j] = wmma_bf16(af[i].v, bf_[j].v, acc[i][j]);
    }

    for (int i = 0; i < 4; ++i) {
        for (int j = 0; j < 4; ++j) {
            int col = n0 + j * 16 + lm;
            float bv = bias ? bias[col] : 0.f;
            int rowb = m0 + i * 16 + lh * 8;
            for (int r = 0; r < 8; ++r) {
                float vres = acc[i][j][r] + bv;
                size_t o = (size_t)(rowb + r) * ldc + col;
                if (Cf) Cf[o] = vres;
                if (Cb) Cb[o] = f2bf(vres);
            }
        }
    }
}

// ---------------- bf16 WMMA GEMM, 32x32 per wave (small-N path, e.g. N=64) ----------------
__global__ __launch_bounds__(256)
void k_gemm_bf16(const unsigned short* __restrict__ A, int lda,
                 const unsigned short* __restrict__ Bt, int ldb,
                 const float* __restrict__ bias,
                 float* __restrict__ Cf, unsigned short* __restrict__ Cb,
                 int ldc, int K) {
    const int lane = threadIdx.x & 31;
    const int w    = threadIdx.x >> 5;
    const int lm   = lane & 15;
    const int lh   = lane >> 4;
    const int m0   = blockIdx.x * 128 + (w & 3) * 32;
    const int n0   = blockIdx.y * 64  + (w >> 2) * 32;

    const v8f vz = {0.f,0.f,0.f,0.f,0.f,0.f,0.f,0.f};
    v8f acc00 = vz, acc01 = vz, acc10 = vz, acc11 = vz;

    for (int k0 = 0; k0 < K; k0 += 32) {
        F16x16 a0, a1, b0, b1;
        const u32x4* p;
        p = (const u32x4*)(A + (size_t)(m0 + lm) * lda + k0 + lh * 8);
        a0.u[0] = p[0]; a0.u[1] = p[2];
        p = (const u32x4*)(A + (size_t)(m0 + 16 + lm) * lda + k0 + lh * 8);
        a1.u[0] = p[0]; a1.u[1] = p[2];
        p = (const u32x4*)(Bt + (size_t)(n0 + lm) * ldb + k0 + lh * 8);
        b0.u[0] = p[0]; b0.u[1] = p[2];
        p = (const u32x4*)(Bt + (size_t)(n0 + 16 + lm) * ldb + k0 + lh * 8);
        b1.u[0] = p[0]; b1.u[1] = p[2];

        acc00 = wmma_bf16(a0.v, b0.v, acc00);
        acc01 = wmma_bf16(a0.v, b1.v, acc01);
        acc10 = wmma_bf16(a1.v, b0.v, acc10);
        acc11 = wmma_bf16(a1.v, b1.v, acc11);
    }

    v8f accs[2][2] = {{acc00, acc01}, {acc10, acc11}};
    for (int i = 0; i < 2; ++i) {
        for (int j = 0; j < 2; ++j) {
            int col = n0 + j * 16 + lm;
            float bv = bias ? bias[col] : 0.f;
            int rowb = m0 + i * 16 + lh * 8;
            for (int r = 0; r < 8; ++r) {
                float vres = accs[i][j][r] + bv;
                size_t o = (size_t)(rowb + r) * ldc + col;
                if (Cf) Cf[o] = vres;
                if (Cb) Cb[o] = f2bf(vres);
            }
        }
    }
}

// ---------------- build qc / kc with RoPE ----------------
// qcb/kcb : [(b*NH+h)][s][192] bf16
__global__ void k_assemble(const unsigned short* __restrict__ qkvb,
                           const float* __restrict__ posq,
                           const float* __restrict__ posk,
                           unsigned short* __restrict__ qcb,
                           unsigned short* __restrict__ kcb) {
    int d = threadIdx.x;            // 0..191
    int s = blockIdx.x;
    int h = blockIdx.y;
    int b = blockIdx.z;
    size_t row   = (size_t)b * SEQ + s;
    size_t obase = (((size_t)(b * NHD + h)) * SEQ + s) * 192 + d;
    float qv, kv;
    if (d < HD) {
        qv = bf2f(qkvb[row * UPD + h * HD + d]);
        kv = bf2f(qkvb[row * UPD + MDIM + h * HD + d]);
    } else {
        int i = d - HD;             // 0..63
        int j = i & 31;
        float ang = (float)s * __powf(RTHETA, -((float)j) / 32.0f);
        float sn, c;
        __sincosf(ang, &sn, &c);
        const float* pq = posq + row * PDIM + h * PHD;
        float xq = pq[i];
        float rq = (i < 32) ? -pq[i + 32] : pq[i - 32];
        qv = xq * c + rq * sn;
        const float* pk = posk + row * PHD;
        float xk = pk[i];
        float rk = (i < 32) ? -pk[i + 32] : pk[i - 32];
        kv = xk * c + rk * sn;
    }
    qcb[obase] = f2bf(qv);
    kcb[obase] = f2bf(kv);
}

// vT[(b*NH+h)][d(0..127)][s] = V  (so P*V B-fragments are contiguous in key dim)
__global__ void k_transpose_v(const unsigned short* __restrict__ qkvb,
                              unsigned short* __restrict__ vT) {
    size_t idx = (size_t)blockIdx.x * blockDim.x + threadIdx.x;
    if (idx >= (size_t)BATCH * NHD * HD * SEQ) return;
    int s = (int)(idx % SEQ);
    size_t t = idx / SEQ;
    int d = (int)(t % HD);  t /= HD;
    int h = (int)(t % NHD);
    int b = (int)(t / NHD);
    vT[idx] = qkvb[((size_t)b * SEQ + s) * UPD + 2 * MDIM + h * HD + d];
}

// ---------------- causal flash attention (per-wave, 16 q-rows) ----------------
__global__ __launch_bounds__(256)
void k_flash_attn(const unsigned short* __restrict__ qc,
                  const unsigned short* __restrict__ kc,
                  const unsigned short* __restrict__ vT,
                  unsigned short* __restrict__ attnb) {
    __shared__ __attribute__((aligned(32))) unsigned short pbuf[8][16 * 32];
    const int lane = threadIdx.x & 31;
    const int w    = threadIdx.x >> 5;
    const int lm   = lane & 15;
    const int lh   = lane >> 4;
    const int h    = blockIdx.y;
    const int b    = blockIdx.z;
    const int bh   = b * NHD + h;
    const int qrow0 = blockIdx.x * 128 + w * 16;

    const unsigned short* qb = qc + (size_t)bh * SEQ * 192;
    const unsigned short* kb = kc + (size_t)bh * SEQ * 192;
    const unsigned short* vb = vT + (size_t)bh * HD * SEQ;

    // Q fragments for all 6 K-chunks of the 192-dim head, resident in VGPRs
    F16x16 qa[6];
    for (int c = 0; c < 6; ++c) {
        const u32x4* p = (const u32x4*)(qb + (size_t)(qrow0 + lm) * 192 + c * 32 + lh * 8);
        qa[c].u[0] = p[0]; qa[c].u[1] = p[2];
    }

    const v8f vz = {0.f,0.f,0.f,0.f,0.f,0.f,0.f,0.f};
    v8f o[8];
    for (int j = 0; j < 8; ++j) o[j] = vz;
    float mrow[8], lrow[8];
    for (int r = 0; r < 8; ++r) { mrow[r] = -1e30f; lrow[r] = 0.f; }

    const float scale = 0.07216878364870323f;   // 1/sqrt(192)

    for (int k0 = 0; k0 < qrow0 + 16; k0 += 32) {
        // scores: 16x32 = Q(16x192) . K^T
        v8f s0 = vz, s1 = vz;
        for (int c = 0; c < 6; ++c) {
            F16x16 bk0, bk1;
            const u32x4* p;
            p = (const u32x4*)(kb + (size_t)(k0 + lm) * 192 + c * 32 + lh * 8);
            bk0.u[0] = p[0]; bk0.u[1] = p[2];
            p = (const u32x4*)(kb + (size_t)(k0 + 16 + lm) * 192 + c * 32 + lh * 8);
            bk1.u[0] = p[0]; bk1.u[1] = p[2];
            s0 = wmma_bf16(qa[c].v, bk0.v, s0);
            s1 = wmma_bf16(qa[c].v, bk1.v, s1);
        }
        const int key0 = k0 + lm;
        const int key1 = k0 + 16 + lm;
        // online softmax per row (row r lives in lanes 0-15, row r+8 in lanes 16-31)
        for (int r = 0; r < 8; ++r) {
            int qrow = qrow0 + r + lh * 8;
            float x0 = s0[r] * scale; if (key0 > qrow) x0 = -1e30f;
            float x1 = s1[r] * scale; if (key1 > qrow) x1 = -1e30f;
            float mx = fmaxf(x0, x1);
            for (int off = 1; off < 16; off <<= 1) mx = fmaxf(mx, __shfl_xor(mx, off, 32));
            float mnew = fmaxf(mrow[r], mx);
            float corr = __expf(mrow[r] - mnew);
            float p0 = __expf(x0 - mnew);
            float p1 = __expf(x1 - mnew);
            float ps = p0 + p1;
            for (int off = 1; off < 16; off <<= 1) ps += __shfl_xor(ps, off, 32);
            lrow[r] = lrow[r] * corr + ps;
            mrow[r] = mnew;
            for (int j = 0; j < 8; ++j) o[j][r] = o[j][r] * corr;
            int row = r + lh * 8;
            pbuf[w][row * 32 + lm]      = f2bf(p0);
            pbuf[w][row * 32 + 16 + lm] = f2bf(p1);
        }
        // P: D-layout -> A-layout via per-wave LDS bounce (LDS ops are in-order per wave)
        F16x16 pa;
        {
            const u32x4* p = (const u32x4*)&pbuf[w][lm * 32 + lh * 8];
            pa.u[0] = p[0];
            p = (const u32x4*)&pbuf[w][lm * 32 + lh * 8 + 16];
            pa.u[1] = p[0];
        }
        // O += P(16x32) . V(32x128)
        for (int j = 0; j < 8; ++j) {
            F16x16 bv;
            const u32x4* p = (const u32x4*)(vb + (size_t)(j * 16 + lm) * SEQ + k0 + lh * 8);
            bv.u[0] = p[0]; bv.u[1] = p[2];
            o[j] = wmma_bf16(pa.v, bv.v, o[j]);
        }
    }

    // epilogue: normalize and store into [B,S,NH*HD] layout (bf16 for the W_o GEMM)
    for (int r = 0; r < 8; ++r) {
        int qrow = qrow0 + r + lh * 8;
        float inv = 1.0f / lrow[r];
        size_t rowoff = ((size_t)b * SEQ + qrow) * MDIM + h * HD;
        for (int j = 0; j < 8; ++j)
            attnb[rowoff + j * 16 + lm] = f2bf(o[j][r] * inv);
    }
}

// ---------------- host-side orchestration ----------------
extern "C" void kernel_launch(void* const* d_in, const int* in_sizes, int n_in,
                              void* d_out, int out_size, void* d_ws, size_t ws_size,
                              hipStream_t stream) {
    const float* x      = (const float*)d_in[0];
    const float* W_down = (const float*)d_in[1];
    const float* b_down = (const float*)d_in[2];
    const float* W_up   = (const float*)d_in[3];
    const float* b_up   = (const float*)d_in[4];
    const float* W_qpos = (const float*)d_in[5];
    const float* b_qpos = (const float*)d_in[6];
    const float* W_kpos = (const float*)d_in[7];
    const float* b_kpos = (const float*)d_in[8];
    const float* W_o    = (const float*)d_in[9];
    const float* b_o    = (const float*)d_in[10];
    float* out = (float*)d_out;

    char* base = (char*)d_ws;
    size_t off = 0;
    auto alloc = [&](size_t elems, size_t esz) -> void* {
        void* p = base + off;
        off += ((elems * esz + 255) & ~(size_t)255);
        return p;
    };
    unsigned short* xb      = (unsigned short*)alloc((size_t)ROWS * MDIM, 2);
    unsigned short* WdT     = (unsigned short*)alloc((size_t)L3 * MDIM, 2);
    unsigned short* WuT     = (unsigned short*)alloc((size_t)UPD * L3, 2);
    unsigned short* WqT     = (unsigned short*)alloc((size_t)PDIM * LAT, 2);
    unsigned short* WkT     = (unsigned short*)alloc((size_t)PHD * MDIM, 2);
    unsigned short* WoT     = (unsigned short*)alloc((size_t)MDIM * MDIM, 2);
    unsigned short* latentb = (unsigned short*)alloc((size_t)ROWS * L3, 2);
    unsigned short* qkvb    = (unsigned short*)alloc((size_t)ROWS * UPD, 2);
    float*          posq    = (float*)alloc((size_t)ROWS * PDIM, 4);
    float*          posk    = (float*)alloc((size_t)ROWS * PHD, 4);
    unsigned short* qcb     = (unsigned short*)alloc((size_t)BATCH * NHD * SEQ * 192, 2);
    unsigned short* kcb     = (unsigned short*)alloc((size_t)BATCH * NHD * SEQ * 192, 2);
    unsigned short* vTb     = (unsigned short*)alloc((size_t)BATCH * NHD * HD * SEQ, 2);
    unsigned short* attnb   = (unsigned short*)alloc((size_t)ROWS * MDIM, 2);

    auto blocks = [](size_t n) { return (unsigned)((n + 255) / 256); };

    // fp32 -> bf16 (activations) and fp32 -> bf16 transposed (weights, B^T layout)
    k_cvt<<<blocks((size_t)ROWS * MDIM), 256, 0, stream>>>(x, xb, (size_t)ROWS * MDIM);
    k_cvt_tr<<<blocks((size_t)MDIM * L3), 256, 0, stream>>>(W_down, WdT, MDIM, L3);
    k_cvt_tr<<<blocks((size_t)L3 * UPD), 256, 0, stream>>>(W_up, WuT, L3, UPD);
    k_cvt_tr<<<blocks((size_t)LAT * PDIM), 256, 0, stream>>>(W_qpos, WqT, LAT, PDIM);
    k_cvt_tr<<<blocks((size_t)MDIM * PHD), 256, 0, stream>>>(W_kpos, WkT, MDIM, PHD);
    k_cvt_tr<<<blocks((size_t)MDIM * MDIM), 256, 0, stream>>>(W_o, WoT, MDIM, MDIM);

    // latent = x @ W_down + b_down  -> bf16
    k_gemm_bf16_big<<<dim3(ROWS / 256, L3 / 128), 256, 0, stream>>>(
        xb, MDIM, WdT, MDIM, b_down, nullptr, latentb, L3, MDIM);
    // qkv = latent @ W_up + b_up -> bf16   (dominant GEMM: 77 GFLOP)
    k_gemm_bf16_big<<<dim3(ROWS / 256, UPD / 128), 256, 0, stream>>>(
        latentb, L3, WuT, L3, b_up, nullptr, qkvb, UPD, L3);
    // pos_q = latent[:, :512] @ W_qpos + b_qpos -> f32 (RoPE in fp32)
    k_gemm_bf16_big<<<dim3(ROWS / 256, PDIM / 128), 256, 0, stream>>>(
        latentb, L3, WqT, LAT, b_qpos, posq, nullptr, PDIM, LAT);
    // pos_k = x @ W_kpos + b_kpos -> f32   (N=64: small-tile kernel)
    k_gemm_bf16<<<dim3(ROWS / 128, PHD / 64), 256, 0, stream>>>(
        xb, MDIM, WkT, MDIM, b_kpos, posk, nullptr, PHD, MDIM);

    // build qc/kc (RoPE + concat) and transposed V
    k_assemble<<<dim3(SEQ, NHD, BATCH), 192, 0, stream>>>(qkvb, posq, posk, qcb, kcb);
    k_transpose_v<<<blocks((size_t)BATCH * NHD * HD * SEQ), 256, 0, stream>>>(qkvb, vTb);

    // causal flash attention
    k_flash_attn<<<dim3(SEQ / 128, NHD, BATCH), 256, 0, stream>>>(qcb, kcb, vTb, attnb);

    // out = attn @ W_o + b_o -> f32 into d_out
    k_gemm_bf16_big<<<dim3(ROWS / 256, MDIM / 128), 256, 0, stream>>>(
        attnb, MDIM, WoT, MDIM, b_o, out, nullptr, MDIM, MDIM);
}
